// MultiHeadAttention_37349035606568
// MI455X (gfx1250) — compile-verified, matrix operand order checked by reading, not run on hardware
//
#include <hip/hip_runtime.h>

// ---------------------------------------------------------------------------
// Multi-head causal attention for MI455X (gfx1250, wave32, WMMA bf16).
//   B=2, T=2048, C=1024, H=16, d=64
// Phase 0: one-time fp32 -> bf16 conversion of x and Wq/Wk/Wv (workspace).
// Phase 1: Q/K/V projections (bf16 WMMA, fp32 accumulate, ping-pong
//          double-buffered K loop), Q pre-scaled by C^-0.5,
//          Q,K stored [B,H,T,d], V stored transposed [B,H,d,T].
// Phase 2: flash attention, one wave per 16-row Q tile, online softmax fp32,
//          V fragments prefetched across the softmax VALU section.
// ---------------------------------------------------------------------------

typedef __attribute__((ext_vector_type(16))) __bf16 v16bf;
typedef __attribute__((ext_vector_type(8)))  __bf16 v8bf;
typedef __attribute__((ext_vector_type(8)))  float  v8f;

#define BB 2
#define TT 2048
#define CC 1024
#define HH 16
#define DD 64

static __device__ __forceinline__ __bf16 f2bf(float f) {
  union { float f; unsigned u; } v; v.f = f;
  unsigned r = v.u + 0x7FFFu + ((v.u >> 16) & 1u);   // round-to-nearest-even
  unsigned short h = (unsigned short)(r >> 16);
  return __builtin_bit_cast(__bf16, h);
}

// A/B fragment layout (16x32 bf16, per-lane row along K):
//   lane L holds row (L&15); K = base..base+7 in halves 0..7 and
//   K = 16+base..16+base+7 in halves 8..15, with base = 8*(L>=16).
// Caller passes p already offset by base; we read p[0..7] and p[16..23].
static __device__ __forceinline__ v16bf frag_from_bf16(const __bf16* __restrict__ p) {
  v8bf lo = *(const v8bf*)(p);
  v8bf hi = *(const v8bf*)(p + 16);
  return __builtin_shufflevector(lo, hi, 0, 1, 2, 3, 4, 5, 6, 7,
                                         8, 9, 10, 11, 12, 13, 14, 15);
}

static __device__ __forceinline__ v8f wmma_bf16(v16bf a, v16bf b, v8f c) {
  return __builtin_amdgcn_wmma_f32_16x16x32_bf16(false, a, false, b,
                                                 (short)0, c, false, false);
}

// ---------------------------------------------------------------------------
// Phase 0: one-time fp32 -> bf16 conversion (8 elements / thread).
// ---------------------------------------------------------------------------
__global__ __launch_bounds__(256) void f32_to_bf16_kernel(
    const float* __restrict__ src, __bf16* __restrict__ dst, int n)
{
  const int i = (blockIdx.x * blockDim.x + threadIdx.x) * 8;
  if (i >= n) return;
  const float4 a = *(const float4*)(src + i);
  const float4 b = *(const float4*)(src + i + 4);
  v8bf r;
  r[0] = f2bf(a.x); r[1] = f2bf(a.y); r[2] = f2bf(a.z); r[3] = f2bf(a.w);
  r[4] = f2bf(b.x); r[5] = f2bf(b.y); r[6] = f2bf(b.z); r[7] = f2bf(b.w);
  *(v8bf*)(dst + i) = r;
}

// ---------------------------------------------------------------------------
// Phase 1: qkv projection.  One wave per (mat, 16-row M tile, 64-col N strip).
// out[m][n] = sum_k x[m][k] * W[n][k]   (i.e. x @ W.T), all-bf16 operands.
// K loop unrolled by 2 with ping-pong fragment sets: each half-step's loads
// are issued ~10 loads ahead of the WMMAs that consume them, with no
// fragment register copies.
// ---------------------------------------------------------------------------
__global__ __launch_bounds__(256) void qkv_proj_kernel(
    const __bf16* __restrict__ xbf, const __bf16* __restrict__ wbf,
    __bf16* __restrict__ qb, __bf16* __restrict__ kb, __bf16* __restrict__ vtb)
{
  const int tid  = blockIdx.x * blockDim.x + threadIdx.x;
  const int wave = tid >> 5;
  const int lane = tid & 31;

  const int tiles_per_mat = (BB * TT / 16) * (CC / 64);   // 256 * 16 = 4096
  const int mat = wave / tiles_per_mat;                   // 0=Q, 1=K, 2=V
  const int rem = wave % tiles_per_mat;
  const int m0  = (rem / (CC / 64)) * 16;
  const int n0  = (rem % (CC / 64)) * 64;

  const __bf16* __restrict__ W = wbf + (size_t)mat * CC * CC;
  const int lh  = lane & 15;
  const int hig = lane >> 4;
  const int kb8 = hig * 8;

  const __bf16* arow = xbf + (size_t)(m0 + lh) * CC + kb8;
  const __bf16* brow[4];
#pragma unroll
  for (int t = 0; t < 4; ++t)
    brow[t] = W + (size_t)(n0 + t * 16 + lh) * CC + kb8;

  v8f acc[4] = {v8f{}, v8f{}, v8f{}, v8f{}};

  // prologue: set0 holds fragments for k = 0
  v16bf a0 = frag_from_bf16(arow);
  v16bf b0[4];
#pragma unroll
  for (int t = 0; t < 4; ++t) b0[t] = frag_from_bf16(brow[t]);
  v16bf a1, b1[4];

  int k0 = 0;
#pragma unroll 1
  for (; k0 < CC - 64; k0 += 64) {
    // load set1 @ k0+32, then compute with set0 @ k0
    a1 = frag_from_bf16(arow + k0 + 32);
#pragma unroll
    for (int t = 0; t < 4; ++t) b1[t] = frag_from_bf16(brow[t] + k0 + 32);
#pragma unroll
    for (int t = 0; t < 4; ++t) acc[t] = wmma_bf16(a0, b0[t], acc[t]);

    // load set0 @ k0+64, then compute with set1 @ k0+32
    a0 = frag_from_bf16(arow + k0 + 64);
#pragma unroll
    for (int t = 0; t < 4; ++t) b0[t] = frag_from_bf16(brow[t] + k0 + 64);
#pragma unroll
    for (int t = 0; t < 4; ++t) acc[t] = wmma_bf16(a1, b1[t], acc[t]);
  }
  // epilogue: k0 == CC-64; set0 already loaded for k0, load set1 @ k0+32
  a1 = frag_from_bf16(arow + k0 + 32);
#pragma unroll
  for (int t = 0; t < 4; ++t) b1[t] = frag_from_bf16(brow[t] + k0 + 32);
#pragma unroll
  for (int t = 0; t < 4; ++t) acc[t] = wmma_bf16(a0, b0[t], acc[t]);
#pragma unroll
  for (int t = 0; t < 4; ++t) acc[t] = wmma_bf16(a1, b1[t], acc[t]);

  const float scale = (mat == 0) ? 0.03125f : 1.0f;   // fold C^-0.5 into Q

  // D layout: VGPR e holds row m0+e (lanes 0-15) / m0+e+8 (lanes 16-31),
  // column n0 + tile*16 + (lane&15).
#pragma unroll
  for (int e = 0; e < 8; ++e) {
    const int m  = m0 + e + 8 * hig;
    const int bI = m >> 11;          // batch  (T = 2048)
    const int tI = m & 2047;         // time
#pragma unroll
    for (int t = 0; t < 4; ++t) {
      const int n  = n0 + t * 16 + lh;
      const int hI = n >> 6;         // head   (d = 64)
      const int dI = n & 63;
      const __bf16 v = f2bf(acc[t][e] * scale);
      if (mat == 0)
        qb[(((size_t)(bI * HH + hI) * TT) + tI) * DD + dI] = v;
      else if (mat == 1)
        kb[(((size_t)(bI * HH + hI) * TT) + tI) * DD + dI] = v;
      else
        vtb[(((size_t)(bI * HH + hI) * DD) + dI) * TT + tI] = v;
    }
  }
}

// ---------------------------------------------------------------------------
// Phase 2: causal flash attention.  One wave per (b, h, 16-row Q tile).
// ---------------------------------------------------------------------------
__global__ __launch_bounds__(256) void attn_kernel(
    const __bf16* __restrict__ qb, const __bf16* __restrict__ kb,
    const __bf16* __restrict__ vtb, float* __restrict__ out)
{
  __shared__ __align__(16) __bf16 plds[8 * 16 * 32];   // 1KB per wave P tile

  const int tid  = blockIdx.x * blockDim.x + threadIdx.x;
  const int wave = tid >> 5;
  const int lane = tid & 31;
  const int lh   = lane & 15;
  const int hig  = lane >> 4;
  const int kb8  = hig * 8;

  const int qt = wave & 127;          // T/16 = 128 q-tiles
  const int h  = (wave >> 7) & 15;
  const int b  = wave >> 11;
  const int i0 = qt * 16;

  const __bf16* qbase = qb  + ((size_t)(b * HH + h) * TT) * DD;
  const __bf16* kbase = kb  + ((size_t)(b * HH + h) * TT) * DD;
  const __bf16* vbase = vtb + ((size_t)(b * HH + h) * DD) * TT;

  // Q fragments (row i0+lh along d); Q was pre-scaled by C^-0.5.
  const __bf16* qrow = qbase + (size_t)(i0 + lh) * DD + kb8;
  const v16bf aq0 = frag_from_bf16(qrow);        // d = 0..31
  const v16bf aq1 = frag_from_bf16(qrow + 32);   // d = 32..63

  v8f o[4] = {v8f{}, v8f{}, v8f{}, v8f{}};
  float mrow[8], lrow[8];
#pragma unroll
  for (int e = 0; e < 8; ++e) { mrow[e] = -1e30f; lrow[e] = 0.0f; }

  __bf16* pw = plds + (threadIdx.x >> 5) * (16 * 32);

  const int nb = (i0 + 47) >> 5;     // #32-wide K/V blocks covering j <= i0+15
#pragma unroll 1
  for (int jb = 0; jb < nb; ++jb) {
    const int j0 = jb << 5;

    // ---- issue K loads (needed first), then V loads (consumed last) ----
    const __bf16* kr0 = kbase + (size_t)(j0 + lh) * DD + kb8;
    const __bf16* kr1 = kr0 + 16 * DD;
    const v16bf bk00 = frag_from_bf16(kr0);
    const v16bf bk01 = frag_from_bf16(kr0 + 32);
    const v16bf bk10 = frag_from_bf16(kr1);
    const v16bf bk11 = frag_from_bf16(kr1 + 32);

    v16bf bv[4];
#pragma unroll
    for (int t = 0; t < 4; ++t) {
      const __bf16* vr = vbase + (size_t)(t * 16 + lh) * TT + j0 + kb8;
      bv[t] = frag_from_bf16(vr);   // in-flight across S-WMMAs + softmax
    }

    // ---- S = Q K^T  (two 16x16 tiles over 32 keys) ----
    v8f s0 = {}, s1 = {};
    s0 = wmma_bf16(aq0, bk00, s0);
    s0 = wmma_bf16(aq1, bk01, s0);
    s1 = wmma_bf16(aq0, bk10, s1);
    s1 = wmma_bf16(aq1, bk11, s1);

    // ---- causal mask: only the final block crosses the diagonal ----
    if (jb == nb - 1) {
#pragma unroll
      for (int e = 0; e < 8; ++e) {
        const int i = i0 + e + 8 * hig;
        const int j = j0 + lh;
        if (j > i)      s0[e] = -1e30f;
        if (j + 16 > i) s1[e] = -1e30f;
      }
    }

    // ---- online softmax (rows live across 16 lanes within a half) ----
#pragma unroll
    for (int e = 0; e < 8; ++e) {
      float mx = fmaxf(s0[e], s1[e]);
#pragma unroll
      for (int off = 1; off < 16; off <<= 1)
        mx = fmaxf(mx, __shfl_xor(mx, off, 32));
      const float mnew = fmaxf(mrow[e], mx);
      const float sc   = __expf(mrow[e] - mnew);
      mrow[e] = mnew;
      const float p0 = __expf(s0[e] - mnew);
      const float p1 = __expf(s1[e] - mnew);
      float ls = p0 + p1;
#pragma unroll
      for (int off = 1; off < 16; off <<= 1)
        ls += __shfl_xor(ls, off, 32);
      lrow[e] = lrow[e] * sc + ls;
      o[0][e] *= sc; o[1][e] *= sc; o[2][e] *= sc; o[3][e] *= sc;
      s0[e] = p0; s1[e] = p1;
    }

    // ---- P: C-layout -> A-layout via per-wave LDS staging ----
#pragma unroll
    for (int e = 0; e < 8; ++e) {
      const int row = e + 8 * hig;
      pw[row * 32 + lh]      = f2bf(s0[e]);
      pw[row * 32 + 16 + lh] = f2bf(s1[e]);
    }
    asm volatile("s_wait_dscnt 0" ::: "memory");
    const v16bf ap = frag_from_bf16(pw + lh * 32 + kb8);

    // ---- O += P @ V  (V fragments prefetched above) ----
#pragma unroll
    for (int t = 0; t < 4; ++t)
      o[t] = wmma_bf16(ap, bv[t], o[t]);
  }

  // ---- normalize by row sum and store fp32 [B,T,C] ----
#pragma unroll
  for (int e = 0; e < 8; ++e) {
    const float inv = 1.0f / lrow[e];
    const int i = i0 + e + 8 * hig;
    float* orow = out + ((size_t)b * TT + i) * CC + h * DD + lh;
#pragma unroll
    for (int t = 0; t < 4; ++t)
      orow[t * 16] = o[t][e] * inv;
  }
}

// ---------------------------------------------------------------------------
extern "C" void kernel_launch(void* const* d_in, const int* in_sizes, int n_in,
                              void* d_out, int out_size, void* d_ws, size_t ws_size,
                              hipStream_t stream) {
  const float* x  = (const float*)d_in[0];
  const float* Wq = (const float*)d_in[1];
  const float* Wk = (const float*)d_in[2];
  const float* Wv = (const float*)d_in[3];
  float* out = (float*)d_out;

  const size_t qkv_elems = (size_t)BB * HH * TT * DD;     // 4M per tensor
  const size_t x_elems   = (size_t)BB * TT * CC;          // 4M
  const size_t w_elems   = (size_t)CC * CC;               // 1M per weight
  const size_t total     = 3 * qkv_elems + x_elems + 3 * w_elems;
  if (ws_size < total * sizeof(unsigned short)) return;

  __bf16* qb  = (__bf16*)d_ws;
  __bf16* kbp = qb + qkv_elems;
  __bf16* vtb = kbp + qkv_elems;
  __bf16* xbf = vtb + qkv_elems;
  __bf16* wbf = xbf + x_elems;          // Wq | Wk | Wv, contiguous

  // Phase 0: one-time fp32 -> bf16 conversion.
  f32_to_bf16_kernel<<<(int)(x_elems / 8 / 256), 256, 0, stream>>>(x, xbf, (int)x_elems);
  f32_to_bf16_kernel<<<(int)(w_elems / 8 / 256), 256, 0, stream>>>(Wq, wbf,               (int)w_elems);
  f32_to_bf16_kernel<<<(int)(w_elems / 8 / 256), 256, 0, stream>>>(Wk, wbf + w_elems,     (int)w_elems);
  f32_to_bf16_kernel<<<(int)(w_elems / 8 / 256), 256, 0, stream>>>(Wv, wbf + 2 * w_elems, (int)w_elems);

  // Phase 1: projections.
  const int waves_a = 3 * (BB * TT / 16) * (CC / 64);     // 12288 waves
  qkv_proj_kernel<<<waves_a * 32 / 256, 256, 0, stream>>>(xbf, wbf, qb, kbp, vtb);

  // Phase 2: attention.
  const int waves_b = BB * HH * (TT / 16);                // 4096 waves
  attn_kernel<<<waves_b * 32 / 256, 256, 0, stream>>>(qb, kbp, vtb, out);
}
